// Block_9698036154897
// MI455X (gfx1250) — matrix operand order repaired and checked by reading, not compiled, not run on hardware
//
#include <hip/hip_runtime.h>
#include <hip/hip_bf16.h>

// ---------------------------------------------------------------------------
// Transformer encoder block for MI455X (gfx1250), bf16 WMMA compute path.
// ---------------------------------------------------------------------------

typedef __bf16 bf16;
typedef __attribute__((ext_vector_type(16))) __bf16 bf16x16;
typedef __attribute__((ext_vector_type(8)))  __bf16 bf16x8;
typedef __attribute__((ext_vector_type(8)))  float  floatx8;

static constexpr int BATCH  = 8;
static constexpr int SEQ    = 1024;
static constexpr int DMODEL = 768;
static constexpr int NH     = 12;
static constexpr int DHEAD  = 64;
static constexpr int DMLP   = 3072;
static constexpr int ROWS   = BATCH * SEQ;      // 8192
static constexpr int MASK_Q = 512;              // PATCHES//2

// ---------------------------------------------------------------------------
// WMMA fragment helpers (layouts per CDNA5 ISA §7.12.2)
// ---------------------------------------------------------------------------

__device__ __forceinline__ bf16x16 combine8(bf16x8 lo, bf16x8 hi) {
    bf16x16 r;
#pragma unroll
    for (int i = 0; i < 8; ++i) { r[i] = lo[i]; r[i + 8] = hi[i]; }
    return r;
}

// A fragment 16x32 bf16 from row-major [M][ldk].
// lane group g = lane>>4: elems 0..7 -> K = k0+g*8..+7, elems 8..15 -> K = k0+16+g*8..+7
__device__ __forceinline__ bf16x16 load_a_frag(const bf16* __restrict__ A, int ldk,
                                               int row, int k0, int lane) {
    const int grp = lane >> 4;
    const bf16* base = A + (size_t)row * ldk + k0 + grp * 8;
    bf16x8 lo = *(const bf16x8*)(base);
    bf16x8 hi = *(const bf16x8*)(base + 16);
    return combine8(lo, hi);
}

// B fragment 32x16 bf16 from BT row-major [N][ldk] (i.e. B transposed).
// lane: n = n0 + (lane&15), elems 0..7 -> K = k0+g*8..+7, elems 8..15 -> K = k0+16+g*8..+7
__device__ __forceinline__ bf16x16 load_b_frag(const bf16* __restrict__ BT, int ldk,
                                               int n0, int k0, int lane) {
    const int grp = lane >> 4;
    const int n = n0 + (lane & 15);
    const bf16* base = BT + (size_t)n * ldk + k0 + grp * 8;
    bf16x8 lo = *(const bf16x8*)(base);
    bf16x8 hi = *(const bf16x8*)(base + 16);
    return combine8(lo, hi);
}

__device__ __forceinline__ floatx8 wmma_bf16(bf16x16 a, bf16x16 b, floatx8 c) {
    return __builtin_amdgcn_wmma_f32_16x16x32_bf16(false, a, false, b, (short)0, c,
                                                   false, false);
}

// reduce across the 16 lanes of a half-wave (xor masks stay within group)
__device__ __forceinline__ float rowmax16(float v) {
#pragma unroll
    for (int m = 1; m < 16; m <<= 1) v = fmaxf(v, __shfl_xor(v, m, 32));
    return v;
}
__device__ __forceinline__ float rowsum16(float v) {
#pragma unroll
    for (int m = 1; m < 16; m <<= 1) v += __shfl_xor(v, m, 32);
    return v;
}

// ---------------------------------------------------------------------------
// fp32 [K][N] -> bf16 transposed [N][K]
// ---------------------------------------------------------------------------
__global__ __launch_bounds__(256) void transpose_to_bf16(const float* __restrict__ W,
                                                         bf16* __restrict__ WT,
                                                         int K, int N) {
    int idx = blockIdx.x * 256 + threadIdx.x;
    if (idx >= K * N) return;
    int k = idx / N;
    int n = idx - k * N;
    WT[(size_t)n * K + k] = (bf16)W[idx];
}

// ---------------------------------------------------------------------------
// Row LayerNorm: fp32 [ROWS][768] -> bf16, one block (256 thr) per row
// ---------------------------------------------------------------------------
__global__ __launch_bounds__(256) void layernorm_to_bf16(const float* __restrict__ x,
                                                         const float* __restrict__ g,
                                                         const float* __restrict__ bb,
                                                         bf16* __restrict__ out) {
    const int row = blockIdx.x;
    const float* xr = x + (size_t)row * DMODEL;
    __shared__ float ssum[8], ssq[8];
    float v[3];
    float s = 0.f, sq = 0.f;
#pragma unroll
    for (int i = 0; i < 3; ++i) {
        v[i] = xr[threadIdx.x + i * 256];
        s += v[i];
        sq += v[i] * v[i];
    }
#pragma unroll
    for (int m = 1; m < 32; m <<= 1) {
        s += __shfl_xor(s, m, 32);
        sq += __shfl_xor(sq, m, 32);
    }
    const int w = threadIdx.x >> 5;
    if ((threadIdx.x & 31) == 0) { ssum[w] = s; ssq[w] = sq; }
    __syncthreads();
    s = 0.f; sq = 0.f;
#pragma unroll
    for (int i = 0; i < 8; ++i) { s += ssum[i]; sq += ssq[i]; }
    const float mu  = s * (1.0f / DMODEL);
    const float var = sq * (1.0f / DMODEL) - mu * mu;
    const float rs  = rsqrtf(var + 1e-6f);
#pragma unroll
    for (int i = 0; i < 3; ++i) {
        int c = threadIdx.x + i * 256;
        out[(size_t)row * DMODEL + c] = (bf16)((v[i] - mu) * rs * g[c] + bb[c]);
    }
}

// ---------------------------------------------------------------------------
// Tiled WMMA GEMM: C[M][N] = A[M][K] * BT[N][K]^T + bias (+resid) (+relu)
// block = 256 threads = 8 waves (2 M-waves x 4 N-waves); wave tile 64M x 64N
// block tile 128M x 256N. 16 WMMAs per K-step per wave, software-pipelined
// (double-buffered A/B fragments) so loads for step k+1 fly under step k's
// WMMA burst.
// OUT_MODE: 0 = f32 [M][N], 1 = bf16 [M][N], 2 = bf16 V-transposed [B][H][DH][S]
// ---------------------------------------------------------------------------
template <int OUT_MODE, bool RELU, bool HAS_RESID>
__global__ __launch_bounds__(256) void gemm_bf16(const bf16* __restrict__ A,
                                                 const bf16* __restrict__ BT,
                                                 const float* __restrict__ bias,
                                                 const float* __restrict__ resid,
                                                 void* __restrict__ outp,
                                                 int M, int N, int K) {
    const int lane = threadIdx.x & 31;
    const int wid  = __builtin_amdgcn_readfirstlane(threadIdx.x >> 5);  // scalar wave id
    const int ln   = lane & 15;
    const int grp  = lane >> 4;
    const int m0   = blockIdx.y * 128 + (wid & 1) * 64;
    const int n0   = blockIdx.x * 256 + (wid >> 1) * 64;

    floatx8 acc[4][4] = {};
    constexpr int PF = 128;   // B-stream prefetch distance (4 K-steps)

    // prologue: fragments for k = 0
    bf16x16 af[4], bfr[4];
#pragma unroll
    for (int mf = 0; mf < 4; ++mf)
        af[mf] = load_a_frag(A, K, m0 + mf * 16 + ln, 0, lane);
#pragma unroll
    for (int nf = 0; nf < 4; ++nf)
        bfr[nf] = load_b_frag(BT, K, n0 + nf * 16, 0, lane);

    for (int k0 = 0; k0 < K; k0 += 32) {
        const int kn = k0 + 32;
        bf16x16 afn[4], bfn[4];
        if (kn < K) {   // scalar condition: issue next step's loads early
#pragma unroll
            for (int mf = 0; mf < 4; ++mf)
                afn[mf] = load_a_frag(A, K, m0 + mf * 16 + ln, kn, lane);
#pragma unroll
            for (int nf = 0; nf < 4; ++nf)
                bfn[nf] = load_b_frag(BT, K, n0 + nf * 16, kn, lane);
#pragma unroll
            for (int nf = 0; nf < 4; ++nf)
                __builtin_prefetch(BT + (size_t)(n0 + nf * 16 + ln) * K + k0 + PF, 0, 1);
        }
        // 16 WMMAs on the current (already resident) fragments
#pragma unroll
        for (int nf = 0; nf < 4; ++nf)
#pragma unroll
            for (int mf = 0; mf < 4; ++mf)
                acc[mf][nf] = wmma_bf16(af[mf], bfr[nf], acc[mf][nf]);
#pragma unroll
        for (int mf = 0; mf < 4; ++mf) af[mf] = afn[mf];
#pragma unroll
        for (int nf = 0; nf < 4; ++nf) bfr[nf] = bfn[nf];
    }

#pragma unroll
    for (int nf = 0; nf < 4; ++nf) {
        const int col = n0 + nf * 16 + ln;
        const float bv = bias[col];
#pragma unroll
        for (int mf = 0; mf < 4; ++mf) {
#pragma unroll
            for (int r = 0; r < 8; ++r) {
                const int row = m0 + mf * 16 + grp * 8 + r;
                float v = acc[mf][nf][r] + bv;
                if (RELU) v = fmaxf(v, 0.f);
                if (HAS_RESID) v += resid[(size_t)row * N + col];
                if (OUT_MODE == 0) {
                    ((float*)outp)[(size_t)row * N + col] = v;
                } else if (OUT_MODE == 1) {
                    ((bf16*)outp)[(size_t)row * N + col] = (bf16)v;
                } else {
                    // V output transposed: [B][H][DH][S]
                    const int b  = row >> 10;          // row / SEQ
                    const int s  = row & (SEQ - 1);
                    const int h  = col >> 6;           // col / DHEAD
                    const int dh = col & (DHEAD - 1);
                    ((bf16*)outp)[(((size_t)(b * NH + h)) * DHEAD + dh) * SEQ + s] =
                        (bf16)v;
                }
            }
        }
    }
}

// ---------------------------------------------------------------------------
// Flash-style attention, one wave per (b, h, 16-query tile).
// Q,K: bf16 [ROWS][DMODEL] (head offset h*64); VT: bf16 [B][H][DH][S]
// ctx out: bf16 [ROWS][DMODEL]
// Query rows >= MASK_Q have scores zeroed pre-softmax -> uniform probs:
// that tile reduces to ctx = mean(V) (P == ones, divisor 1024).
// Tile coordinates go through readfirstlane so `masked` is a provably scalar
// branch: EXEC stays all-1s around every WMMA.
// ---------------------------------------------------------------------------
__global__ __launch_bounds__(256) void attention_kernel(const bf16* __restrict__ Q,
                                                        const bf16* __restrict__ Km,
                                                        const bf16* __restrict__ VT,
                                                        bf16* __restrict__ ctx) {
    __shared__ __align__(16) bf16 Plds[8][512];   // per-wave 16x32 P tile

    const int w    = threadIdx.x >> 5;
    const int lane = threadIdx.x & 31;
    const int ln   = lane & 15;
    const int grp  = lane >> 4;

    const int idx = __builtin_amdgcn_readfirstlane((int)(blockIdx.x * 8 + w));
    const int qt  = idx & 63;
    const int h   = (idx >> 6) % NH;
    const int b   = idx / (64 * NH);
    const int q0  = qt * 16;

    const bf16* Qb = Q  + (size_t)b * SEQ * DMODEL + h * DHEAD;
    const bf16* Kb = Km + (size_t)b * SEQ * DMODEL + h * DHEAD;
    const bf16* Vb = VT + ((size_t)(b * NH + h)) * DHEAD * SEQ;
    const bool masked = (q0 >= MASK_Q);   // scalar

    floatx8 ctxa[4] = {};
    float mrow[8], lrow[8];
#pragma unroll
    for (int r = 0; r < 8; ++r) { mrow[r] = -1e30f; lrow[r] = 0.f; }

    bf16x16 qf0, qf1, ones;
#pragma unroll
    for (int i = 0; i < 16; ++i) ones[i] = (bf16)1.0f;
    if (!masked) {
        qf0 = load_a_frag(Qb, DMODEL, q0 + ln, 0, lane);
        qf1 = load_a_frag(Qb, DMODEL, q0 + ln, 32, lane);
    }

    bf16* pl = &Plds[w][0];

    for (int kc = 0; kc < SEQ / 32; ++kc) {
        const int key0 = kc * 32;
        // issue V-fragment loads first: they fly during the softmax work
        bf16x16 vb[4];
#pragma unroll
        for (int nf = 0; nf < 4; ++nf)
            vb[nf] = load_b_frag(Vb, SEQ, nf * 16, key0, lane);

        if (!masked) {
            floatx8 s0 = {}, s1 = {};
            {
                bf16x16 kb0 = load_b_frag(Kb, DMODEL, key0,      0, lane);
                bf16x16 kb1 = load_b_frag(Kb, DMODEL, key0 + 16, 0, lane);
                s0 = wmma_bf16(qf0, kb0, s0);
                s1 = wmma_bf16(qf0, kb1, s1);
                kb0 = load_b_frag(Kb, DMODEL, key0,      32, lane);
                kb1 = load_b_frag(Kb, DMODEL, key0 + 16, 32, lane);
                s0 = wmma_bf16(qf1, kb0, s0);
                s1 = wmma_bf16(qf1, kb1, s1);
            }
            const float scale = 0.125f;   // 1/sqrt(64)
            floatx8 p0 = {}, p1 = {};
#pragma unroll
            for (int r = 0; r < 8; ++r) {
                const float a = s0[r] * scale;
                const float c = s1[r] * scale;
                float t = rowmax16(fmaxf(a, c));
                const float mn   = fmaxf(mrow[r], t);
                const float corr = __expf(mrow[r] - mn);
                const float e0 = __expf(a - mn);
                const float e1 = __expf(c - mn);
                const float rs = rowsum16(e0 + e1);
                lrow[r] = lrow[r] * corr + rs;
                mrow[r] = mn;
                p0[r] = e0;
                p1[r] = e1;
#pragma unroll
                for (int nf = 0; nf < 4; ++nf) ctxa[nf][r] *= corr;
            }
            // stage P (C-layout f32) -> LDS as bf16 row-major 16x32
#pragma unroll
            for (int r = 0; r < 8; ++r) {
                pl[(grp * 8 + r) * 32 + ln]      = (bf16)p0[r];
                pl[(grp * 8 + r) * 32 + 16 + ln] = (bf16)p1[r];
            }
            // read back in A-fragment layout (same wave: DS is in-order)
            bf16x8 lo = *(const bf16x8*)(pl + ln * 32 + grp * 8);
            bf16x8 hi = *(const bf16x8*)(pl + ln * 32 + 16 + grp * 8);
            bf16x16 pa = combine8(lo, hi);
#pragma unroll
            for (int nf = 0; nf < 4; ++nf)
                ctxa[nf] = wmma_bf16(pa, vb[nf], ctxa[nf]);
        } else {
            // uniform attention: P tile == 1 (no softmax work at all)
#pragma unroll
            for (int nf = 0; nf < 4; ++nf)
                ctxa[nf] = wmma_bf16(ones, vb[nf], ctxa[nf]);
        }
    }

#pragma unroll
    for (int nf = 0; nf < 4; ++nf) {
        const int col = h * DHEAD + nf * 16 + ln;
#pragma unroll
        for (int r = 0; r < 8; ++r) {
            const int row = b * SEQ + q0 + grp * 8 + r;
            const float inv = masked ? (1.0f / SEQ) : (1.0f / lrow[r]);
            ctx[(size_t)row * DMODEL + col] = (bf16)(ctxa[nf][r] * inv);
        }
    }
}

// ---------------------------------------------------------------------------
// Launch
// ---------------------------------------------------------------------------
extern "C" void kernel_launch(void* const* d_in, const int* in_sizes, int n_in,
                              void* d_out, int out_size, void* d_ws, size_t ws_size,
                              hipStream_t stream) {
    const float* x    = (const float*)d_in[0];
    const float* Wq   = (const float*)d_in[1];
    const float* bq   = (const float*)d_in[2];
    const float* Wk   = (const float*)d_in[3];
    const float* bk   = (const float*)d_in[4];
    const float* Wv   = (const float*)d_in[5];
    const float* bv   = (const float*)d_in[6];
    const float* Wo   = (const float*)d_in[7];
    const float* bo   = (const float*)d_in[8];
    const float* ln1g = (const float*)d_in[9];
    const float* ln1b = (const float*)d_in[10];
    const float* ln2g = (const float*)d_in[11];
    const float* ln2b = (const float*)d_in[12];
    const float* W1   = (const float*)d_in[13];
    const float* b1   = (const float*)d_in[14];
    const float* W2   = (const float*)d_in[15];
    const float* b2   = (const float*)d_in[16];

    char* ws = (char*)d_ws;
    size_t off = 0;
    auto alloc = [&](size_t bytes) {
        size_t r = off;
        off += (bytes + 255) & ~(size_t)255;
        return r;
    };
    const size_t AB = (size_t)ROWS * DMODEL * sizeof(bf16);  // activation buf, 12.6 MB

    bf16* WTq = (bf16*)(ws + alloc((size_t)DMODEL * DMODEL * 2));
    bf16* WTk = (bf16*)(ws + alloc((size_t)DMODEL * DMODEL * 2));
    bf16* WTv = (bf16*)(ws + alloc((size_t)DMODEL * DMODEL * 2));
    bf16* WTo = (bf16*)(ws + alloc((size_t)DMODEL * DMODEL * 2));
    bf16* WT1 = (bf16*)(ws + alloc((size_t)DMODEL * DMLP * 2));  // [3072][768]
    bf16* WT2 = (bf16*)(ws + alloc((size_t)DMLP * DMODEL * 2));  // [768][3072]

    // regionA: xn | Q | K | VT  -- dead after attention+Wo, reused as h1 (exactly 4*AB)
    char* regionA = ws + alloc(4 * AB);
    bf16* xn  = (bf16*)(regionA);
    bf16* Qb  = (bf16*)(regionA + AB);
    bf16* Kb  = (bf16*)(regionA + 2 * AB);
    bf16* VTb = (bf16*)(regionA + 3 * AB);
    bf16* h1  = (bf16*)(regionA);                     // [8192][3072] bf16 = 4*AB

    char* regionB = ws + alloc(AB);                   // ctx, later xn2
    bf16* ctx = (bf16*)(regionB);
    bf16* xn2 = (bf16*)(regionB);

    float* x1 = (float*)(ws + alloc((size_t)ROWS * DMODEL * sizeof(float)));

    float* out = (float*)d_out;

    // 1. weights -> bf16 transposed
    auto tgrid = [](int n) { return dim3((n + 255) / 256); };
    transpose_to_bf16<<<tgrid(DMODEL * DMODEL), 256, 0, stream>>>(Wq, WTq, DMODEL, DMODEL);
    transpose_to_bf16<<<tgrid(DMODEL * DMODEL), 256, 0, stream>>>(Wk, WTk, DMODEL, DMODEL);
    transpose_to_bf16<<<tgrid(DMODEL * DMODEL), 256, 0, stream>>>(Wv, WTv, DMODEL, DMODEL);
    transpose_to_bf16<<<tgrid(DMODEL * DMODEL), 256, 0, stream>>>(Wo, WTo, DMODEL, DMODEL);
    transpose_to_bf16<<<tgrid(DMODEL * DMLP),   256, 0, stream>>>(W1, WT1, DMODEL, DMLP);
    transpose_to_bf16<<<tgrid(DMLP * DMODEL),   256, 0, stream>>>(W2, WT2, DMLP, DMODEL);

    // 2. LN1 -> bf16
    layernorm_to_bf16<<<ROWS, 256, 0, stream>>>(x, ln1g, ln1b, xn);

    // 3. Q, K, V projections (V stored transposed [B][H][DH][S])
    dim3 gproj(DMODEL / 256, ROWS / 128);
    gemm_bf16<1, false, false><<<gproj, 256, 0, stream>>>(xn, WTq, bq, nullptr, Qb,
                                                          ROWS, DMODEL, DMODEL);
    gemm_bf16<1, false, false><<<gproj, 256, 0, stream>>>(xn, WTk, bk, nullptr, Kb,
                                                          ROWS, DMODEL, DMODEL);
    gemm_bf16<2, false, false><<<gproj, 256, 0, stream>>>(xn, WTv, bv, nullptr, VTb,
                                                          ROWS, DMODEL, DMODEL);

    // 4. attention -> ctx (bf16 [ROWS][DMODEL])
    attention_kernel<<<BATCH * NH * (SEQ / 16) / 8, 256, 0, stream>>>(Qb, Kb, VTb, ctx);

    // 5. Wo + residual -> x1 (fp32)
    gemm_bf16<0, false, true><<<gproj, 256, 0, stream>>>(ctx, WTo, bo, x, x1,
                                                         ROWS, DMODEL, DMODEL);

    // 6. LN2 -> bf16 (overwrites ctx region; ctx is dead)
    layernorm_to_bf16<<<ROWS, 256, 0, stream>>>(x1, ln2g, ln2b, xn2);

    // 7. MLP1: relu(xn2 @ W1 + b1) -> h1 bf16 (overwrites xn/Q/K/VT region)
    dim3 gmlp1(DMLP / 256, ROWS / 128);
    gemm_bf16<1, true, false><<<gmlp1, 256, 0, stream>>>(xn2, WT1, b1, nullptr, h1,
                                                         ROWS, DMLP, DMODEL);

    // 8. MLP2: h1 @ W2 + b2 + x1 -> out fp32
    gemm_bf16<0, false, true><<<gproj, 256, 0, stream>>>(h1, WT2, b2, x1, out,
                                                         ROWS, DMODEL, DMLP);
}
